// CrossModalAttention_84490596647390
// MI455X (gfx1250) — compile-verified, matrix operand order checked by reading, not run on hardware
//
#include <hip/hip_runtime.h>
#include <hip/hip_bf16.h>

typedef __bf16 bf16;
typedef __attribute__((ext_vector_type(16))) __bf16 v16bf;
typedef __attribute__((ext_vector_type(8)))  __bf16 bf16x8;
typedef __attribute__((ext_vector_type(8)))  float  v8f;
typedef int v4i __attribute__((vector_size(16)));   // matches builtin param type

// ---------- CDNA5 async-to-LDS path (guarded; sync fallback keeps build green) ----
#if defined(__gfx1250__) && __has_builtin(__builtin_amdgcn_global_load_async_to_lds_b128)
#define ASYNC_LDS_OK 1
#endif

// ---------- helpers ----------

__device__ __forceinline__ bf16 f2b(float f) {
    // f32 -> bf16, round-to-nearest-even
    unsigned int u = __builtin_bit_cast(unsigned int, f);
    unsigned int r = (u + 0x7FFFu + ((u >> 16) & 1u)) >> 16;
    return __builtin_bit_cast(bf16, (unsigned short)r);
}

template<typename T> __device__ __forceinline__ T cvt_out(float f);
template<> __device__ __forceinline__ float cvt_out<float>(float f) { return f; }
template<> __device__ __forceinline__ bf16  cvt_out<bf16>(float f)  { return f2b(f); }

// copy 8 bf16 (16B) global -> LDS; async on gfx1250, sync fallback otherwise
__device__ __forceinline__ void copy16(const bf16* g, bf16* l) {
#if defined(ASYNC_LDS_OK)
    __builtin_amdgcn_global_load_async_to_lds_b128(
        (__attribute__((address_space(1))) v4i*)(void*)g,
        (__attribute__((address_space(3))) v4i*)(void*)l, 0, 0);
#else
    *(bf16x8*)l = *(const bf16x8*)g;
#endif
}

__device__ __forceinline__ void wait_async0() {
#if defined(ASYNC_LDS_OK)
#if __has_builtin(__builtin_amdgcn_s_wait_asynccnt)
    __builtin_amdgcn_s_wait_asynccnt(0);
#else
    asm volatile("s_wait_asynccnt 0x0" ::: "memory");
#endif
#endif
}

// 16-bit A/B fragment: lane holds two contiguous 8-element K segments:
//   k = k0 + 16*(e/8) + 8*hi + (e%8),  hi = lane/16
__device__ __forceinline__ v16bf frag_ld(const bf16* base, int hi) {
    bf16x8 lo = *(const bf16x8*)(base + hi * 8);
    bf16x8 hh = *(const bf16x8*)(base + 16 + hi * 8);
    return __builtin_shufflevector(lo, hh, 0,1,2,3,4,5,6,7,8,9,10,11,12,13,14,15);
}

__device__ __forceinline__ v8f wmma_bf16(v16bf a, v16bf b, v8f c) {
    return __builtin_amdgcn_wmma_f32_16x16x32_bf16(
        /*neg_a=*/false, a, /*neg_b=*/false, b,
        /*c_mod=*/(short)0, c, /*reuse_a=*/false, /*reuse_b=*/false);
}

// ---------- constants ----------
#define B_      16
#define HW_     4096
#define CV_     768
#define NL_     32
#define KC_     512     // KEY_C == VAL_C == OUT_C
#define NH_     8
#define DK_     64
#define LDSSTR  40      // padded bf16 row stride (80B, 16B aligned)

// ---------- kernel 0: f32 -> bf16 conversion (x, Wq, Wo) ----------
__global__ __launch_bounds__(256)
void cvt_bf16_kernel(const float* __restrict__ src, bf16* __restrict__ dst, int n) {
    const int i = blockIdx.x * 256 + threadIdx.x;
    if (i < n) dst[i] = f2b(src[i]);
}

// ---------- kernel 1: K/V projection + mask ----------
// k[b,h,l,d] (bf16)   v[b,h,d,l] (bf16)
__global__ __launch_bounds__(256)
void proj_kv_kernel(const float* __restrict__ l, const float* __restrict__ lmask,
                    const float* __restrict__ Wk, const float* __restrict__ bk,
                    const float* __restrict__ Wv, const float* __restrict__ bv,
                    bf16* __restrict__ kout, bf16* __restrict__ vout) {
    const int bh = blockIdx.x;            // 0..127
    const int b  = bh >> 3;
    const int h  = bh & 7;
    for (int i = 0; i < 8; ++i) {
        const int idx = threadIdx.x + i * 256;   // 0..2047
        const int d   = idx & 63;
        const int li  = idx >> 6;                // 0..31
        const int c   = h * DK_ + d;
        const float* wkr = Wk + (size_t)c * CV_;
        const float* wvr = Wv + (size_t)c * CV_;
        const float* lb  = l + (size_t)b * CV_ * NL_ + li;
        float ka = 0.f, va = 0.f;
        for (int cl = 0; cl < CV_; ++cl) {
            const float lv = lb[(size_t)cl * NL_];
            ka += wkr[cl] * lv;
            va += wvr[cl] * lv;
        }
        const float m = lmask[b * NL_ + li];
        ka = (ka + bk[c]) * m;
        va = (va + bv[c]) * m;
        kout[((size_t)bh * NL_ + li) * DK_ + d] = f2b(ka);
        vout[((size_t)bh * DK_ + d) * NL_ + li] = f2b(va);
    }
}

// ---------- kernel 2/5: WMMA GEMM  C[M,512] = A[M,K] * Bw[512,K]^T + bias ----------
// A, Bw in bf16. Block tile 128x64, 8 waves as 4(M)x2(N), each wave 2x2 WMMA tiles.
// Double-buffered LDS staging via async global->LDS copies (ASYNCcnt).
__device__ __forceinline__ void stage_tiles(const bf16* __restrict__ A,
                                            const bf16* __restrict__ Bw, int K,
                                            size_t blockM, int blockN, int k0,
                                            bf16* As, bf16* Bs, int tid) {
    // A tile: 128 rows x 32 k  => 512 chunks of 8 bf16 (16B)
    #pragma unroll
    for (int c = 0; c < 2; ++c) {
        const int cid = tid + c * 256;
        const int row = cid >> 2, ko = (cid & 3) * 8;
        copy16(A + (blockM + row) * (size_t)K + k0 + ko, As + row * LDSSTR + ko);
    }
    // B tile: 64 rows(n) x 32 k => 256 chunks
    {
        const int n = tid >> 2, ko = (tid & 3) * 8;
        copy16(Bw + (size_t)(blockN + n) * K + k0 + ko, Bs + n * LDSSTR + ko);
    }
}

__global__ __launch_bounds__(256)
void gemm_wmma(const bf16* __restrict__ A, const bf16* __restrict__ Bw,
               const float* __restrict__ bias, float* __restrict__ C, int K) {
    __shared__ __attribute__((aligned(16))) bf16 As[2][128 * LDSSTR];
    __shared__ __attribute__((aligned(16))) bf16 Bs[2][64 * LDSSTR];

    const int tid   = threadIdx.x;
    const int lane  = tid & 31;
    const int wave  = tid >> 5;
    const int waveM = wave >> 1;          // 0..3
    const int waveN = wave & 1;           // 0..1
    const int hi    = lane >> 4;
    const int ln    = lane & 15;
    const int    blockN = blockIdx.x * 64;            // N fastest: A-tile L2 reuse
    const size_t blockM = (size_t)blockIdx.y * 128;

    v8f acc[2][2] = {};
    const int nk = K >> 5;
    int cur = 0;

    stage_tiles(A, Bw, K, blockM, blockN, 0, As[0], Bs[0], tid);

    for (int i = 0; i < nk; ++i) {
        wait_async0();          // tile i resident (this wave's copies)
        __syncthreads();        // everyone's tile i resident; prev reads done
        if (i + 1 < nk)
            stage_tiles(A, Bw, K, blockM, blockN, (i + 1) * 32,
                        As[cur ^ 1], Bs[cur ^ 1], tid);

        v16bf a0 = frag_ld(&As[cur][(waveM * 32 + ln) * LDSSTR], hi);
        v16bf a1 = frag_ld(&As[cur][(waveM * 32 + 16 + ln) * LDSSTR], hi);
        v16bf b0 = frag_ld(&Bs[cur][(waveN * 32 + ln) * LDSSTR], hi);
        v16bf b1 = frag_ld(&Bs[cur][(waveN * 32 + 16 + ln) * LDSSTR], hi);

        acc[0][0] = wmma_bf16(a0, b0, acc[0][0]);
        acc[0][1] = wmma_bf16(a0, b1, acc[0][1]);
        acc[1][0] = wmma_bf16(a1, b0, acc[1][0]);
        acc[1][1] = wmma_bf16(a1, b1, acc[1][1]);

        __syncthreads();        // all reads of tile i done before overwrite
        cur ^= 1;
    }

    // epilogue: + bias, f32 store. C layout: row = r + 8*hi, col = lane%16
    #pragma unroll
    for (int mt = 0; mt < 2; ++mt) {
        #pragma unroll
        for (int nt = 0; nt < 2; ++nt) {
            const int coln = blockN + waveN * 32 + nt * 16 + ln;
            const float bv = bias[coln];
            #pragma unroll
            for (int r = 0; r < 8; ++r) {
                const size_t grow = blockM + waveM * 32 + mt * 16 + r + hi * 8;
                C[grow * KC_ + coln] = acc[mt][nt][r] + bv;
            }
        }
    }
}

// ---------- kernel 3a/6a: instance-norm stats over HW per (b,c) ----------
__global__ __launch_bounds__(256)
void inorm_stats(const float* __restrict__ src, float* __restrict__ mean,
                 float* __restrict__ rstd) {
    const int b = blockIdx.x >> 9;
    const int c = blockIdx.x & 511;
    __shared__ float s1[256], s2[256];
    float a = 0.f, q = 0.f;
    for (int i = threadIdx.x; i < HW_; i += 256) {
        const float v = src[((size_t)b * HW_ + i) * KC_ + c];
        a += v; q += v * v;
    }
    s1[threadIdx.x] = a; s2[threadIdx.x] = q;
    __syncthreads();
    for (int s = 128; s > 0; s >>= 1) {
        if (threadIdx.x < s) { s1[threadIdx.x] += s1[threadIdx.x + s];
                               s2[threadIdx.x] += s2[threadIdx.x + s]; }
        __syncthreads();
    }
    if (threadIdx.x == 0) {
        const float mu  = s1[0] * (1.f / HW_);
        const float var = s2[0] * (1.f / HW_) - mu * mu;
        mean[blockIdx.x] = mu;
        rstd[blockIdx.x] = rsqrtf(var + 1e-5f);
    }
}

// ---------- kernel 3b/6b: apply instance-norm ----------
template<typename T>
__global__ __launch_bounds__(256)
void inorm_apply(const float* __restrict__ src, const float* __restrict__ mean,
                 const float* __restrict__ rstd, T* __restrict__ dst) {
    const size_t idx = (size_t)blockIdx.x * 256 + threadIdx.x;
    const int c  = (int)(idx & 511);
    const int b  = (int)(idx >> 9) >> 12;      // row = b*4096 + hw
    const int bc = b * KC_ + c;
    dst[idx] = cvt_out<T>((src[idx] - mean[bc]) * rstd[bc]);
}

// ---------- kernel 4: fused attention (per-wave 16-row tile) ----------
__global__ __launch_bounds__(128)
void attn_kernel(const bf16* __restrict__ qn, const bf16* __restrict__ kmat,
                 const bf16* __restrict__ vmat, const float* __restrict__ lmask,
                 bf16* __restrict__ attn_out, float* __restrict__ sim_temp) {
    __shared__ __attribute__((aligned(16))) bf16 ldsP[4][16 * LDSSTR];

    const int tid  = threadIdx.x;
    const int lane = tid & 31;
    const int wv   = tid >> 5;
    const int hi   = lane >> 4;
    const int ln   = lane & 15;

    const int gw   = blockIdx.x * 4 + wv;      // 0..32767
    const int b    = gw >> 11;                 // /(8*256)
    const int rem  = gw & 2047;
    const int h    = rem >> 8;
    const int mt   = rem & 255;
    const int row0 = mt * 16;
    const int bh   = b * NH_ + h;

    // --- sim = (q * K) * scale + mask_bias ---
    const bf16* qrow = qn + ((size_t)(b * HW_ + row0 + ln)) * KC_ + h * DK_;
    const v16bf aq0 = frag_ld(qrow, hi);        // d = 0..31
    const v16bf aq1 = frag_ld(qrow + 32, hi);   // d = 32..63

    const bf16* kb0 = kmat + ((size_t)bh * NL_ + ln) * DK_;       // l = ln
    const bf16* kb1 = kmat + ((size_t)bh * NL_ + 16 + ln) * DK_;  // l = 16+ln
    const v16bf bk00 = frag_ld(kb0,      hi);
    const v16bf bk01 = frag_ld(kb0 + 32, hi);
    const v16bf bk10 = frag_ld(kb1,      hi);
    const v16bf bk11 = frag_ld(kb1 + 32, hi);

    v8f c0 = {}, c1 = {};
    c0 = wmma_bf16(aq0, bk00, c0);
    c0 = wmma_bf16(aq1, bk01, c0);
    c1 = wmma_bf16(aq0, bk10, c1);
    c1 = wmma_bf16(aq1, bk11, c1);

    const float scale = 0.04419417382415922f;   // 512^-0.5
    const float bias0 = 10000.f * lmask[b * NL_ + ln]      - 10000.f;
    const float bias1 = 10000.f * lmask[b * NL_ + 16 + ln] - 10000.f;

    // --- softmax over 32 keys per row; rows live across 16-lane groups ---
    #pragma unroll
    for (int r = 0; r < 8; ++r) {
        const float s0 = c0[r] * scale + bias0;
        const float s1 = c1[r] * scale + bias1;
        float t = fmaxf(s0, s1);
        t = fmaxf(t, __shfl_xor(t, 1));
        t = fmaxf(t, __shfl_xor(t, 2));
        t = fmaxf(t, __shfl_xor(t, 4));
        t = fmaxf(t, __shfl_xor(t, 8));
        const float e0 = __expf(s0 - t);
        const float e1 = __expf(s1 - t);
        float sm = e0 + e1;
        sm += __shfl_xor(sm, 1);
        sm += __shfl_xor(sm, 2);
        sm += __shfl_xor(sm, 4);
        sm += __shfl_xor(sm, 8);
        const float inv = 1.f / sm;
        const float p0 = e0 * inv, p1 = e1 * inv;
        // sim_temp = sum over l of softmax(sim)
        float st = p0 + p1;
        st += __shfl_xor(st, 1);
        st += __shfl_xor(st, 2);
        st += __shfl_xor(st, 4);
        st += __shfl_xor(st, 8);
        const int mrow = r + hi * 8;
        if (ln == 0)
            sim_temp[(size_t)bh * HW_ + row0 + mrow] = st;
        ldsP[wv][mrow * LDSSTR + ln]      = f2b(p0);
        ldsP[wv][mrow * LDSSTR + 16 + ln] = f2b(p1);
    }
    __syncthreads();

    // --- out = P * V^T : A = P (16x32), B[n=d][k=l] from v[b,h,d,l] ---
    const v16bf ap = frag_ld(&ldsP[wv][ln * LDSSTR], hi);
    v8f o[4];
    #pragma unroll
    for (int nt = 0; nt < 4; ++nt) {
        const bf16* vb = vmat + ((size_t)bh * DK_ + nt * 16 + ln) * NL_;
        const v16bf bvv = frag_ld(vb, hi);
        v8f z = {};
        o[nt] = wmma_bf16(ap, bvv, z);
    }

    #pragma unroll
    for (int nt = 0; nt < 4; ++nt) {
        const int col = h * DK_ + nt * 16 + ln;
        #pragma unroll
        for (int r = 0; r < 8; ++r) {
            const int mrow = row0 + r + hi * 8;
            attn_out[((size_t)(b * HW_ + mrow)) * KC_ + col] = f2b(o[nt][r]);
        }
    }
}

// ---------- launch ----------
extern "C" void kernel_launch(void* const* d_in, const int* in_sizes, int n_in,
                              void* d_out, int out_size, void* d_ws, size_t ws_size,
                              hipStream_t stream) {
    const float* x     = (const float*)d_in[0];
    const float* l     = (const float*)d_in[1];
    const float* lmask = (const float*)d_in[2];
    const float* Wq    = (const float*)d_in[3];
    const float* bq    = (const float*)d_in[4];
    const float* Wk    = (const float*)d_in[5];
    const float* bk    = (const float*)d_in[6];
    const float* Wv    = (const float*)d_in[7];
    const float* bv    = (const float*)d_in[8];
    const float* Wo    = (const float*)d_in[9];
    const float* bo    = (const float*)d_in[10];

    char* ws = (char*)d_ws;
    // scratch layout (bytes)
    float* qraw  = (float*)(ws);                     // 16*4096*512*4   = 134217728 (reused for o-raw)
    bf16*  qn    = (bf16*) (ws + 134217728);         // 16*4096*512*2   =  67108864
    bf16*  attn  = (bf16*) (ws + 201326592);         // 16*4096*512*2   =  67108864
    bf16*  kmat  = (bf16*) (ws + 268435456);         // 16*8*32*64*2    =    524288
    bf16*  vmat  = (bf16*) (ws + 268959744);         // 16*8*64*32*2    =    524288
    float* mean  = (float*)(ws + 269484032);         // 16*512*4        =     32768
    float* rstd  = (float*)(ws + 269516800);         // 16*512*4        =     32768
    bf16*  xbf   = (bf16*) (ws + 269549568);         // 65536*768*2     = 100663296
    bf16*  wqbf  = (bf16*) (ws + 370212864);         // 512*768*2       =    786432
    bf16*  wobf  = (bf16*) (ws + 370999296);         // 512*512*2       =    524288

    float* out  = (float*)d_out;
    float* simt = out + (size_t)B_ * HW_ * KC_;      // tuple output #2

    // 0. pre-convert GEMM operands to bf16 (async-copyable, halves A traffic)
    cvt_bf16_kernel<<<(65536 * CV_) / 256, 256, 0, stream>>>(x,  xbf,  65536 * CV_);
    cvt_bf16_kernel<<<(KC_ * CV_) / 256,   256, 0, stream>>>(Wq, wqbf, KC_ * CV_);
    cvt_bf16_kernel<<<(KC_ * KC_) / 256,   256, 0, stream>>>(Wo, wobf, KC_ * KC_);

    // 1. K/V projections (small)
    proj_kv_kernel<<<B_ * NH_, 256, 0, stream>>>(l, lmask, Wk, bk, Wv, bv, kmat, vmat);

    // 2. Q projection: [65536,768] x [768,512]
    gemm_wmma<<<dim3(KC_ / 64, 65536 / 128), 256, 0, stream>>>(xbf, wqbf, bq, qraw, CV_);

    // 3. InstanceNorm(q) -> bf16
    inorm_stats<<<B_ * KC_, 256, 0, stream>>>(qraw, mean, rstd);
    inorm_apply<bf16><<<(B_ * HW_ * KC_) / 256, 256, 0, stream>>>(qraw, mean, rstd, qn);

    // 4. fused attention (softmax over 32 keys) -> bf16 attn + sim_temp
    attn_kernel<<<(B_ * NH_ * (HW_ / 16)) / 4, 128, 0, stream>>>(qn, kmat, vmat, lmask, attn, simt);

    // 5. output projection: [65536,512] x [512,512]
    gemm_wmma<<<dim3(KC_ / 64, 65536 / 128), 256, 0, stream>>>(attn, wobf, bo, qraw, KC_);

    // 6. InstanceNorm(out) -> f32 d_out
    inorm_stats<<<B_ * KC_, 256, 0, stream>>>(qraw, mean, rstd);
    inorm_apply<float><<<(B_ * HW_ * KC_) / 256, 256, 0, stream>>>(qraw, mean, rstd, out);
}
